// NASGNN_24266565222462
// MI455X (gfx1250) — matrix operand order repaired and checked by reading, not compiled, not run on hardware
//
#include <hip/hip_runtime.h>
#include <hip/hip_bf16.h>
#include <stdint.h>

#define N_NODES 50000
#define N_EDGES 600000
#define DIM     128
#define N_CLASS 40
#define N_LABEL 25000
#define ROWS_PB 80          // rows per block: 50000 / 80 = 625 exactly; 5 row-tiles of 16

typedef __attribute__((ext_vector_type(16))) __bf16 v16bf;
typedef __attribute__((ext_vector_type(8)))  float  v8f;

// ---------------- degree / dinv ----------------
__global__ void init_deg_kernel(float* deg) {
    int i = blockIdx.x * blockDim.x + threadIdx.x;
    if (i < N_NODES) deg[i] = 1.0f;   // self-loop contributes 1
}

__global__ void accum_deg_kernel(const long long* __restrict__ dst, float* deg) {
    int e = blockIdx.x * blockDim.x + threadIdx.x;
    if (e < N_EDGES) atomicAdd(&deg[(int)dst[e]], 1.0f);
}

__global__ void finish_dinv_kernel(float* dinv) {
    int i = blockIdx.x * blockDim.x + threadIdx.x;
    if (i < N_NODES) dinv[i] = rsqrtf(dinv[i]);
}

// ---------------- f32 weights -> pre-packed bf16 WMMA B-fragments ----------------
// Packed layout: fragment f = (coltile w)*4 + kt, then lane 0..31, then 16 contiguous
// bf16 per lane.  B k order chosen here (contiguous): lane half 0 -> K=kt*32+0..15, half 1 -> +16.
__global__ void cvt_w_kernel(const float* __restrict__ W, __bf16* __restrict__ Wpk) {
    int t = blockIdx.x * blockDim.x + threadIdx.x;   // 0..1023
    if (t >= 1024) return;
    const int f    = t >> 5;            // 0..31: w = f>>2, kt = f&3
    const int lane = t & 31;
    const int w    = f >> 2;
    const int kt   = f & 3;
    const int half = lane >> 4;
    const int n    = w * 16 + (lane & 15);
    #pragma unroll
    for (int i = 0; i < 16; ++i) {
        const int K = kt * 32 + half * 16 + i;
        Wpk[t * 16 + i] = (__bf16)W[K * DIM + n];
    }
}

// ---------------- WMMA GEMM + fused self-loop epilogue ----------------
// act = (relu0? relu(in0) : in0) + relu(in1) + relu(in2)   (in1/in2 optional)
// h   = act @ W                       (kept for the edge gather)
// out = h * dinv^2 + beta * bias      (self-loop + bias init, scatter adds on top)
// Block = 256 threads (8 waves), computes ROWS_PB x 128.  Wave w owns columns
// [16w,16w+16) and keeps its 4 B-fragments in registers across 5 row-tiles.
__global__ __launch_bounds__(256)
void gemm_wmma_kernel(const float* __restrict__ in0,
                      const float* __restrict__ in1,
                      const float* __restrict__ in2,
                      int relu0,
                      const __bf16* __restrict__ Wpk,
                      const float* __restrict__ dinv,
                      const float* __restrict__ bias,
                      float beta,
                      float* __restrict__ h,
                      float* __restrict__ out) {
    __shared__ __bf16 As[ROWS_PB][DIM + 8];   // +8 bf16 pad to spread LDS banks

    const int t    = threadIdx.x;
    const int row0 = blockIdx.x * ROWS_PB;

    // cooperative load + fused relu-sum + f32->bf16: 2560 float4 / 256 thr = 10 iters
    #pragma unroll
    for (int it = 0; it < (ROWS_PB * DIM / 4) / 256; ++it) {
        const int vid = it * 256 + t;
        const int r   = vid >> 5;            // 32 float4 per row
        const int c   = (vid & 31) * 4;
        const size_t gi = (size_t)(row0 + r) * DIM + c;
        float4 v = *(const float4*)(in0 + gi);
        if (relu0) {
            v.x = fmaxf(v.x, 0.f); v.y = fmaxf(v.y, 0.f);
            v.z = fmaxf(v.z, 0.f); v.w = fmaxf(v.w, 0.f);
        }
        if (in1) {
            float4 u = *(const float4*)(in1 + gi);
            v.x += fmaxf(u.x, 0.f); v.y += fmaxf(u.y, 0.f);
            v.z += fmaxf(u.z, 0.f); v.w += fmaxf(u.w, 0.f);
        }
        if (in2) {
            float4 u = *(const float4*)(in2 + gi);
            v.x += fmaxf(u.x, 0.f); v.y += fmaxf(u.y, 0.f);
            v.z += fmaxf(u.z, 0.f); v.w += fmaxf(u.w, 0.f);
        }
        As[r][c + 0] = (__bf16)v.x;  As[r][c + 1] = (__bf16)v.y;
        As[r][c + 2] = (__bf16)v.z;  As[r][c + 3] = (__bf16)v.w;
    }
    __syncthreads();

    const int wave = t >> 5;
    const int lane = t & 31;
    const int nloc = lane & 15;
    const int half = lane >> 4;
    const int ncol = wave * 16 + nloc;
    const float bcol = beta * bias[ncol];

    // B fragments: packed contiguous, 2x global_load_b128 per fragment
    const v16bf* Bp = (const v16bf*)Wpk;
    v16bf bf[4];
    #pragma unroll
    for (int kt = 0; kt < 4; ++kt) bf[kt] = Bp[(wave * 4 + kt) * 32 + lane];

    #pragma unroll
    for (int rt = 0; rt < ROWS_PB / 16; ++rt) {
        const int m = rt * 16 + nloc;        // A row within LDS tile
        v8f acc = {};
        #pragma unroll
        for (int kt = 0; kt < 4; ++kt) {
            v16bf a;
            // A k order used here (contiguous, matches packed B): half=0 -> K 0..15, half=1 -> K 16..31
            #pragma unroll
            for (int i = 0; i < 8; ++i) {
                a[i]     = As[m][kt * 32 + half * 16 + i];
                a[i + 8] = As[m][kt * 32 + half * 16 + 8 + i];
            }
            acc = __builtin_amdgcn_wmma_f32_16x16x32_bf16(
                      false, a, false, bf[kt], (short)0, acc, false, false);
        }
        // C/D layout: VGPR v -> M = v + 8*half, N = lane&15
        #pragma unroll
        for (int v = 0; v < 8; ++v) {
            const int row = row0 + rt * 16 + v + 8 * half;
            const float val = acc[v];
            const float di  = dinv[row];
            h  [(size_t)row * DIM + ncol] = val;
            out[(size_t)row * DIM + ncol] = val * di * di + bcol;
        }
    }
}

// ---------------- edge scatter: out[dst] += h[src] * dinv[s]*dinv[d] ----------------
// One wave per edge; each lane handles 4 consecutive floats (32*4 = 128).
__global__ __launch_bounds__(256)
void scatter_kernel(const long long* __restrict__ src,
                    const long long* __restrict__ dst,
                    const float* __restrict__ dinv,
                    const float* __restrict__ h,
                    float* __restrict__ out) {
    int e    = (blockIdx.x * blockDim.x + threadIdx.x) >> 5;
    int lane = threadIdx.x & 31;
    if (e >= N_EDGES) return;
    if (e + 4096 < N_EDGES) {                 // warm L2/L0 for upcoming edge ids
        __builtin_prefetch(src + e + 4096, 0, 0);
        __builtin_prefetch(dst + e + 4096, 0, 0);
    }
    int s = (int)src[e], d = (int)dst[e];
    float w = dinv[s] * dinv[d];
    const float4* hrow = (const float4*)(h + (size_t)s * DIM);
    float4 v = hrow[lane];
    float* o = out + (size_t)d * DIM + lane * 4;
    atomicAdd(o + 0, v.x * w);
    atomicAdd(o + 1, v.y * w);
    atomicAdd(o + 2, v.z * w);
    atomicAdd(o + 3, v.w * w);
}

// ---------------- fused log_softmax + head, labeled rows only ----------------
__global__ __launch_bounds__(128)
void head_kernel(const float* __restrict__ c3,
                 const long long* __restrict__ idx,
                 const float* __restrict__ Wh,
                 const float* __restrict__ bh,
                 float* __restrict__ out) {
    __shared__ float row[DIM];
    __shared__ float red[DIM];
    const int m = blockIdx.x;
    const int j = threadIdx.x;
    const long long n = idx[m];
    const float v = c3[n * DIM + j];

    red[j] = v;
    __syncthreads();
    for (int s = 64; s > 0; s >>= 1) {
        if (j < s) red[j] = fmaxf(red[j], red[j + s]);
        __syncthreads();
    }
    const float mx = red[0];
    __syncthreads();

    red[j] = __expf(v - mx);
    __syncthreads();
    for (int s = 64; s > 0; s >>= 1) {
        if (j < s) red[j] += red[j + s];
        __syncthreads();
    }
    const float lse = __logf(red[0]) + mx;
    row[j] = v - lse;
    __syncthreads();

    if (j < N_CLASS) {
        float acc = bh[j];
        #pragma unroll 8
        for (int k = 0; k < DIM; ++k) acc += row[k] * Wh[k * N_CLASS + j];
        out[(size_t)m * N_CLASS + j] = acc;
    }
}

// ---------------- launch ----------------
extern "C" void kernel_launch(void* const* d_in, const int* in_sizes, int n_in,
                              void* d_out, int out_size, void* d_ws, size_t ws_size,
                              hipStream_t stream) {
    (void)in_sizes; (void)n_in; (void)out_size; (void)ws_size;

    const float*     x    = (const float*)d_in[0];
    const long long* ei   = (const long long*)d_in[1];   // int64 [2, E]
    const long long* src  = ei;
    const long long* dst  = ei + N_EDGES;
    const long long* nli  = (const long long*)d_in[2];   // int64 [M]
    const float* W0 = (const float*)d_in[3];  const float* b0 = (const float*)d_in[4];
    const float* W1 = (const float*)d_in[5];  const float* b1 = (const float*)d_in[6];
    const float* W2 = (const float*)d_in[7];  const float* b2 = (const float*)d_in[8];
    const float* W3 = (const float*)d_in[9];  const float* b3 = (const float*)d_in[10];
    const float* Wh = (const float*)d_in[11]; const float* bh = (const float*)d_in[12];

    const size_t FEAT_BYTES = (size_t)N_NODES * DIM * sizeof(float);
    char* p = (char*)d_ws;
    auto take = [&](size_t bytes) { char* q = p; p += (bytes + 255) & ~(size_t)255; return q; };
    float*  dinv = (float*)take(sizeof(float) * N_NODES);
    __bf16* Wpk  = (__bf16*)take(sizeof(unsigned short) * DIM * DIM * 4);
    float*  h    = (float*)take(FEAT_BYTES);
    float*  c0   = (float*)take(FEAT_BYTES);
    float*  c1   = (float*)take(FEAT_BYTES);
    float*  c2   = (float*)take(FEAT_BYTES);
    float*  c3   = (float*)take(FEAT_BYTES);

    const int gN  = (N_NODES + 255) / 256;
    const int gE  = (N_EDGES + 255) / 256;
    const int gMM = N_NODES / ROWS_PB;            // 625 (exact)
    const int gSC = (N_EDGES * 32 + 255) / 256;   // 75000

    // degrees -> dinv
    init_deg_kernel   <<<gN, 256, 0, stream>>>(dinv);
    accum_deg_kernel  <<<gE, 256, 0, stream>>>(dst, dinv);
    finish_dinv_kernel<<<gN, 256, 0, stream>>>(dinv);

    // weights -> packed bf16 fragments
    const float* Ws[4] = {W0, W1, W2, W3};
    for (int k = 0; k < 4; ++k)
        cvt_w_kernel<<<4, 256, 0, stream>>>(Ws[k], Wpk + (size_t)k * DIM * DIM);

    // pass 0: c0 = A_hat (x W0) + b0
    gemm_wmma_kernel<<<gMM, 256, 0, stream>>>(x,  nullptr, nullptr, 0,
                                              Wpk + 0 * DIM * DIM, dinv, b0, 1.0f, h, c0);
    scatter_kernel  <<<gSC, 256, 0, stream>>>(src, dst, dinv, h, c0);

    // pass 1: c1 = A_hat (relu(c0) W1) + b1
    gemm_wmma_kernel<<<gMM, 256, 0, stream>>>(c0, nullptr, nullptr, 1,
                                              Wpk + 1 * DIM * DIM, dinv, b1, 1.0f, h, c1);
    scatter_kernel  <<<gSC, 256, 0, stream>>>(src, dst, dinv, h, c1);

    // pass 2 (fused): c2 = A_hat ((relu(c0)+relu(c1)) W2) + 2 b2
    gemm_wmma_kernel<<<gMM, 256, 0, stream>>>(c0, c1, nullptr, 1,
                                              Wpk + 2 * DIM * DIM, dinv, b2, 2.0f, h, c2);
    scatter_kernel  <<<gSC, 256, 0, stream>>>(src, dst, dinv, h, c2);

    // pass 3 (fused): c3 = A_hat ((relu(c0)+relu(c1)+relu(c2)) W3) + 3 b3
    gemm_wmma_kernel<<<gMM, 256, 0, stream>>>(c0, c1, c2, 1,
                                              Wpk + 3 * DIM * DIM, dinv, b3, 3.0f, h, c3);
    scatter_kernel  <<<gSC, 256, 0, stream>>>(src, dst, dinv, h, c3);

    // head: pred = log_softmax(c3[idx]) @ Wh + bh  (labeled rows only)
    head_kernel<<<N_LABEL, 128, 0, stream>>>(c3, nli, Wh, bh, (float*)d_out);
}
